// Generator_31293131719467
// MI455X (gfx1250) — matrix operand order repaired and checked by reading, not compiled, not run on hardware
//
#include <hip/hip_runtime.h>

// ---------------------------------------------------------------------------
// CDNA5 (gfx1250) wave32 WMMA transformer-imputation pipeline.
// All heavy GEMMs use v_wmma_f32_16x16x32_f16 (f16 in, f32 accumulate).
// Attention is flash-style (no O(T^2) materialization), FFN is fused in LDS.
// Tile staging uses GLOBAL_LOAD_ASYNC_TO_LDS_B128 (ASYNCcnt) when available.
// ---------------------------------------------------------------------------

typedef _Float16 half_t;
typedef _Float16 v16h __attribute__((ext_vector_type(16)));
typedef float    v8f  __attribute__((ext_vector_type(8)));
typedef unsigned int v8u __attribute__((ext_vector_type(8)));
typedef int v4i_async __attribute__((vector_size(16)));  // builtin's pointee type

#define WMMA_F16(a, b, c) \
  __builtin_amdgcn_wmma_f32_16x16x32_f16(false, (a), false, (b), (short)0, (c), false, false)

#if defined(__has_builtin)
#if __has_builtin(__builtin_amdgcn_global_load_async_to_lds_b128) && \
    __has_builtin(__builtin_amdgcn_s_wait_asynccnt)
#define HAVE_ASYNC_LDS 1
#endif
#endif
#ifndef HAVE_ASYNC_LDS
#define HAVE_ASYNC_LDS 0
#endif

// 16-byte global -> LDS copy. Async (ASYNCcnt-tracked, no VGPR round trip)
// on gfx1250 toolchains that expose the builtin; synchronous fallback else.
// Builtin signature (from clang diagnostics): (v4i __device__* src_global,
// v4i __shared__* dst_lds, imm offset, imm cpol).
__device__ __forceinline__ void copy16_g2l(const void* g, void* l) {
#if HAVE_ASYNC_LDS
  __builtin_amdgcn_global_load_async_to_lds_b128(
      (__attribute__((address_space(1))) v4i_async*)g,
      (__attribute__((address_space(3))) v4i_async*)l, 0, 0);
#else
  unsigned int tmp[4];
  __builtin_memcpy(tmp, g, 16);
  __builtin_memcpy(l, tmp, 16);
#endif
}
__device__ __forceinline__ void copy16_wait() {
#if HAVE_ASYNC_LDS
  __builtin_amdgcn_s_wait_asynccnt(0);
#endif
}

__device__ __forceinline__ unsigned int ldu32(const half_t* p) {
  unsigned int v; __builtin_memcpy(&v, p, 4); return v;
}
__device__ __forceinline__ void stu32(half_t* p, unsigned int v) {
  __builtin_memcpy(p, &v, 4);
}
__device__ __forceinline__ unsigned int pack2h(float a, float b) {
  half_t h[2]; h[0] = (half_t)a; h[1] = (half_t)b;
  unsigned int v; __builtin_memcpy(&v, h, 4); return v;
}

// A fragment (16x32 f16). Lane holds row = lane%16. Per CDNA5 ISA layout:
//   lo lanes: VGPR0..3 -> K=0..7, VGPR4..7 -> K=16..23 ; hi lanes: +8.
__device__ __forceinline__ v16h load_frag_a(const half_t* rowp, int hi) {
  v8u u; const int hb = hi * 8;
#pragma unroll
  for (int v = 0; v < 8; ++v) {
    const int kk = ((v < 4) ? (2 * v) : (2 * v + 8)) + hb;
    u[v] = ldu32(rowp + kk);
  }
  return __builtin_bit_cast(v16h, u);
}
// B fragment (32x16 f16) from an [n][k] (k-contiguous) tile.
// Lane holds col = lane%16, same K order as A: lo lanes K=0..7,16..23 ; hi lanes +8.
__device__ __forceinline__ v16h load_frag_b(const half_t* colp, int hi) {
  v8u u;
#pragma unroll
  for (int v = 0; v < 8; ++v) u[v] = ldu32(colp + ((v < 4) ? (2 * v) : (2 * v + 8)) + hi * 8);
  return __builtin_bit_cast(v16h, u);
}

__device__ __forceinline__ float red_max16(float v) {
  v = fmaxf(v, __shfl_xor(v, 1)); v = fmaxf(v, __shfl_xor(v, 2));
  v = fmaxf(v, __shfl_xor(v, 4)); v = fmaxf(v, __shfl_xor(v, 8));
  return v;
}
__device__ __forceinline__ float red_sum16(float v) {
  v += __shfl_xor(v, 1); v += __shfl_xor(v, 2);
  v += __shfl_xor(v, 4); v += __shfl_xor(v, 8);
  return v;
}
__device__ __forceinline__ float red_sum32(float v) {
  v = red_sum16(v); v += __shfl_xor(v, 16); return v;
}
__device__ __forceinline__ float sigmoidf_(float x) {
  return 1.0f / (1.0f + __expf(-x));
}

// ---------------------------------------------------------------------------
// f32 -> f16 weight conversion
// ---------------------------------------------------------------------------
__global__ __launch_bounds__(256) void k_cvt(const float* __restrict__ s,
                                             half_t* __restrict__ d, int n) {
  int i = blockIdx.x * 256 + threadIdx.x;
  if (i < n) d[i] = (half_t)s[i];
}

// ---------------------------------------------------------------------------
// Stage 1 embedding: xe = x*w1x + b1x + PE ; tde = exp(-relu(td*w1m + b1m))
// row r = t*128 + b*16 + f   (T, B*F, E) layout, E inner.
// ---------------------------------------------------------------------------
__global__ __launch_bounds__(256) void k_embed1(
    const float* __restrict__ x, const float* __restrict__ td,
    const float* __restrict__ w1x, const float* __restrict__ b1x,
    const float* __restrict__ w1m, const float* __restrict__ b1m,
    float* __restrict__ xe, half_t* __restrict__ tde) {
  size_t idx = (size_t)blockIdx.x * 256 + threadIdx.x;
  int e = (int)(idx & 255);
  size_t r = idx >> 8;
  int t = (int)(r >> 7);
  int bf = (int)(r & 127);
  int b = bf >> 4, f = bf & 15;
  size_t gi = ((size_t)b * 512 + t) * 16 + f;
  int e2 = e & ~1;
  float div = __expf((float)e2 * (-9.210340371976184f / 256.0f));
  float ang = (float)t * div;
  float pe = (e & 1) ? __cosf(ang) : __sinf(ang);
  xe[idx] = x[gi] * w1x[e] + b1x[e] + pe;
  float tv = td[gi] * w1m[e] + b1m[e];
  tde[idx] = (half_t)__expf(-fmaxf(tv, 0.0f));
}

// ---------------------------------------------------------------------------
// Generic WMMA GEMM: C[N,M] = A[N,K] @ W[M,K]^T + bias.
// Templated on A precision (branch-free inner loop). 64x64 tile / block,
// 8 waves, each wave owns a 16x32 slab (two 16x16 accumulators).
// LDS tile pitch = 40 halves (80 B) so every 8-half chunk is 16B aligned
// (async b128 friendly) and the 20-dword lane stride is bank-conflict-free.
// N % 64 == 0, M % 64 == 0, K % 32 == 0 for all launches here.
// ---------------------------------------------------------------------------
template <int AF16>
__global__ __launch_bounds__(256) void k_gemm(
    const void* __restrict__ Av, const half_t* __restrict__ W,
    const float* __restrict__ bias, float* __restrict__ C32,
    half_t* __restrict__ C16, int Kd, int M) {
  __shared__ half_t Al[64 * 40];
  __shared__ half_t Bl[64 * 40];
  const int tid = threadIdx.x;
  const int lane = tid & 31, wave = tid >> 5;
  const int l16 = lane & 15, hi = lane >> 4;
  const int n0 = blockIdx.x * 64;
  const int m0 = blockIdx.y * 64;
  const int wr = wave >> 1, wc = wave & 1;
  const int ro = wr * 16, co = wc * 32;
  const int srow = tid >> 2;       // 0..63
  const int skb = (tid & 3) * 8;   // 0,8,16,24

  v8f c0 = {}; v8f c1 = {};
  for (int kt = 0; kt < Kd; kt += 32) {
    if (AF16) {
      const half_t* src = (const half_t*)Av + (size_t)(n0 + srow) * Kd + kt + skb;
      copy16_g2l(src, &Al[srow * 40 + skb]);
    } else {
      const float* src = (const float*)Av + (size_t)(n0 + srow) * Kd + kt + skb;
#pragma unroll
      for (int j = 0; j < 4; ++j)
        stu32(&Al[srow * 40 + skb + 2 * j], pack2h(src[2 * j], src[2 * j + 1]));
    }
    copy16_g2l(W + (size_t)(m0 + srow) * Kd + kt + skb, &Bl[srow * 40 + skb]);
    copy16_wait();
    __syncthreads();
    v16h a  = load_frag_a(&Al[(ro + l16) * 40], hi);
    v16h b0 = load_frag_b(&Bl[(co + l16) * 40], hi);
    v16h b1 = load_frag_b(&Bl[(co + 16 + l16) * 40], hi);
    c0 = WMMA_F16(a, b0, c0);
    c1 = WMMA_F16(a, b1, c1);
    __syncthreads();
  }
  const int colg0 = m0 + co + l16;
  const int colg1 = colg0 + 16;
  const float bb0 = bias ? bias[colg0] : 0.0f;
  const float bb1 = bias ? bias[colg1] : 0.0f;
#pragma unroll
  for (int i = 0; i < 8; ++i) {
    int rowg = n0 + ro + i + hi * 8;
    float v0 = c0[i] + bb0, v1 = c1[i] + bb1;
    if (C32) {
      C32[(size_t)rowg * M + colg0] = v0;
      C32[(size_t)rowg * M + colg1] = v1;
    }
    if (C16) {
      C16[(size_t)rowg * M + colg0] = (half_t)v0;
      C16[(size_t)rowg * M + colg1] = (half_t)v1;
    }
  }
}

// ---------------------------------------------------------------------------
// Flash attention: one block per (batch, head). hd = 32, T = 512.
// Q/K/V/O rows: r = t*B1 + batch, col = head*32 + d (f16, stride 256).
// K,V async-staged in LDS; per 16-row Q tile, online softmax over 32-key
// chunks: 2 WMMAs for S = Q@K^T (K-dim = hd = 32), 2 WMMAs for O += P@V.
// ---------------------------------------------------------------------------
__global__ __launch_bounds__(256) void k_attn(
    const half_t* __restrict__ Q, const half_t* __restrict__ K,
    const half_t* __restrict__ V, half_t* __restrict__ O, int B1) {
  extern __shared__ half_t smem[];
  half_t* Kl = smem;                 // [512][40]
  half_t* Vl = smem + 512 * 40;      // [512][40]
  half_t* Pl = smem + 2 * 512 * 40;  // [8 waves][16*32]
  const int tid = threadIdx.x, lane = tid & 31, wave = tid >> 5;
  const int l16 = lane & 15, hi = lane >> 4;
  const int b = blockIdx.x / 8, h = blockIdx.x % 8;
  const size_t colbase = (size_t)h * 32;
  const float scale = 0.1767766952966369f;  // 1/sqrt(32)

  for (int idx = tid; idx < 2048; idx += 256) {  // 2048 16B chunks per matrix
    int t = idx >> 2;
    int c8 = (idx & 3) * 8;
    size_t g = ((size_t)t * B1 + b) * 256 + colbase + c8;
    copy16_g2l(&K[g], &Kl[t * 40 + c8]);
    copy16_g2l(&V[g], &Vl[t * 40 + c8]);
  }
  copy16_wait();
  __syncthreads();
  half_t* Pw = Pl + wave * 512;

  for (int qt0 = wave * 16; qt0 < 512; qt0 += 128) {
    v8u qu;
    {
      const half_t* rowq = Q + ((size_t)(qt0 + l16) * B1 + b) * 256 + colbase;
      const int hb = hi * 8;
#pragma unroll
      for (int v = 0; v < 8; ++v) {
        const int kk = ((v < 4) ? (2 * v) : (2 * v + 8)) + hb;
        qu[v] = ldu32(rowq + kk);
      }
    }
    v16h qa = __builtin_bit_cast(v16h, qu);
    float mrow[8], lrow[8];
    v8f o0 = {}; v8f o1 = {};
#pragma unroll
    for (int i = 0; i < 8; ++i) { mrow[i] = -3.0e38f; lrow[i] = 0.0f; }

    for (int c0 = 0; c0 < 512; c0 += 32) {
      v16h bk0 = load_frag_b(&Kl[(c0 + l16) * 40], hi);
      v16h bk1 = load_frag_b(&Kl[(c0 + 16 + l16) * 40], hi);
      v8f s0 = {}; v8f s1 = {};
      s0 = WMMA_F16(qa, bk0, s0);
      s1 = WMMA_F16(qa, bk1, s1);
#pragma unroll
      for (int i = 0; i < 8; ++i) {
        float a0 = s0[i] * scale, a1 = s1[i] * scale;
        float mx = red_max16(fmaxf(a0, a1));
        float mn = fmaxf(mrow[i], mx);
        float scl = __expf(mrow[i] - mn);
        float e0 = __expf(a0 - mn), e1 = __expf(a1 - mn);
        float rs = red_sum16(e0 + e1);
        mrow[i] = mn;
        lrow[i] = lrow[i] * scl + rs;
        o0[i] *= scl; o1[i] *= scl;
        Pw[(i + hi * 8) * 32 + l16] = (half_t)e0;
        Pw[(i + hi * 8) * 32 + 16 + l16] = (half_t)e1;
      }
      __asm__ volatile("" ::: "memory");  // per-wave LDS RAW: HW keeps DS in order
      v8u pu;
      {
        const half_t* prow = Pw + l16 * 32;
        const int hb = hi * 8;
#pragma unroll
        for (int v = 0; v < 8; ++v) {
          const int kk = ((v < 4) ? (2 * v) : (2 * v + 8)) + hb;
          pu[v] = ldu32(prow + kk);
        }
      }
      v16h pa = __builtin_bit_cast(v16h, pu);
      v16h bv0, bv1;
#pragma unroll
      for (int hh = 0; hh < 16; ++hh) {
        int key = c0 + ((hh < 8) ? hh : (hh + 8)) + 8 * hi;
        bv0[hh] = Vl[key * 40 + l16];
        bv1[hh] = Vl[key * 40 + 16 + l16];
      }
      o0 = WMMA_F16(pa, bv0, o0);
      o1 = WMMA_F16(pa, bv1, o1);
    }
#pragma unroll
    for (int i = 0; i < 8; ++i) {
      float inv = 1.0f / lrow[i];
      int qrow = qt0 + i + hi * 8;
      size_t g = ((size_t)qrow * B1 + b) * 256 + colbase;
      O[g + l16] = (half_t)(o0[i] * inv);
      O[g + 16 + l16] = (half_t)(o1[i] * inv);
    }
  }
}

// ---------------------------------------------------------------------------
// out = LayerNorm(X + P * td) ; one wave per row of 256.
// ---------------------------------------------------------------------------
__global__ __launch_bounds__(256) void k_res_td_ln(
    const float* __restrict__ X, const float* __restrict__ P,
    const half_t* __restrict__ TD, const float* __restrict__ G,
    const float* __restrict__ Bt, float* __restrict__ Out) {
  const int lane = threadIdx.x & 31, wave = threadIdx.x >> 5;
  const size_t base = ((size_t)blockIdx.x * 8 + wave) * 256;
  float u[8]; float s = 0.0f;
#pragma unroll
  for (int j = 0; j < 8; ++j) {
    int e = lane + 32 * j;
    float tdv = TD ? (float)TD[base + e] : 1.0f;
    float v = X[base + e] + P[base + e] * tdv;
    u[j] = v; s += v;
  }
  float mu = red_sum32(s) * (1.0f / 256.0f);
  float vs = 0.0f;
#pragma unroll
  for (int j = 0; j < 8; ++j) { float d = u[j] - mu; vs += d * d; }
  float rstd = rsqrtf(red_sum32(vs) * (1.0f / 256.0f) + 1e-5f);
#pragma unroll
  for (int j = 0; j < 8; ++j) {
    int e = lane + 32 * j;
    Out[base + e] = (u[j] - mu) * rstd * G[e] + Bt[e];
  }
}

// ---------------------------------------------------------------------------
// Fused FFN: out = LN(in + relu(in@W1^T + b1)@W2^T + b2). 32 rows per block.
// Hidden (32x1024 f16) lives entirely in LDS -> no 268 MB scratch tensor.
// ---------------------------------------------------------------------------
__global__ __launch_bounds__(256) void k_ffn(
    const float* __restrict__ In, const half_t* __restrict__ W1,
    const float* __restrict__ B1v, const half_t* __restrict__ W2,
    const float* __restrict__ B2v, const float* __restrict__ G,
    const float* __restrict__ Bt, float* __restrict__ Out) {
  extern __shared__ char smc[];
  half_t* Al = (half_t*)smc;                              // [32][258]
  half_t* Hl = (half_t*)(smc + 32 * 258 * 2);             // [32][1026]
  float*  Ul = (float*)(smc + 32 * 258 * 2 + 32 * 1026 * 2);  // [32][256]
  const int tid = threadIdx.x, lane = tid & 31, wave = tid >> 5;
  const int l16 = lane & 15, hi = lane >> 4;
  const size_t r0 = (size_t)blockIdx.x * 32;
  const int rg = wave & 1;
  const int cbase = wave >> 1;

  for (int idx = tid; idx < 4096; idx += 256) {  // 32*256 halves as dwords
    int row = idx >> 7;
    int cp = (idx & 127) * 2;
    const float* s = In + (r0 + row) * 256 + cp;
    stu32(&Al[row * 258 + cp], pack2h(s[0], s[1]));
  }
  __syncthreads();

  for (int jj = 0; jj < 16; ++jj) {  // GEMM1: 32x1024, relu
    int cg = cbase + jj * 4;
    v8f c = {};
    for (int kt = 0; kt < 256; kt += 32) {
      v16h a = load_frag_a(&Al[(rg * 16 + l16) * 258 + kt], hi);
      v16h b = load_frag_b(&W1[(size_t)(cg * 16 + l16) * 256 + kt], hi);
      c = WMMA_F16(a, b, c);
    }
#pragma unroll
    for (int i = 0; i < 8; ++i) {
      int rr = rg * 16 + i + hi * 8;
      int cc = cg * 16 + l16;
      Hl[rr * 1026 + cc] = (half_t)fmaxf(c[i] + B1v[cc], 0.0f);
    }
  }
  __syncthreads();

  for (int jj = 0; jj < 4; ++jj) {  // GEMM2: 32x256 + residual
    int cg = cbase + jj * 4;
    v8f c = {};
    for (int kt = 0; kt < 1024; kt += 32) {
      v16h a = load_frag_a(&Hl[(rg * 16 + l16) * 1026 + kt], hi);
      v16h b = load_frag_b(&W2[(size_t)(cg * 16 + l16) * 1024 + kt], hi);
      c = WMMA_F16(a, b, c);
    }
#pragma unroll
    for (int i = 0; i < 8; ++i) {
      int rr = rg * 16 + i + hi * 8;
      int cc = cg * 16 + l16;
      Ul[rr * 256 + cc] = In[(r0 + rr) * 256 + cc] + c[i] + B2v[cc];
    }
  }
  __syncthreads();

  for (int rr = wave; rr < 32; rr += 8) {  // fused LayerNorm
    float u[8]; float s = 0.0f;
#pragma unroll
    for (int j = 0; j < 8; ++j) { u[j] = Ul[rr * 256 + lane + 32 * j]; s += u[j]; }
    float mu = red_sum32(s) * (1.0f / 256.0f);
    float vs = 0.0f;
#pragma unroll
    for (int j = 0; j < 8; ++j) { float d = u[j] - mu; vs += d * d; }
    float rstd = rsqrtf(red_sum32(vs) * (1.0f / 256.0f) + 1e-5f);
#pragma unroll
    for (int j = 0; j < 8; ++j) {
      int e = lane + 32 * j;
      Out[(r0 + rr) * 256 + e] = (u[j] - mu) * rstd * G[e] + Bt[e];
    }
  }
}

// ---------------------------------------------------------------------------
// x_hat = sigmoid(rep @ winv^T + binv) ; X' = m*x + (1-m)*x_hat. One wave/row.
// ---------------------------------------------------------------------------
__global__ __launch_bounds__(256) void k_xprime(
    const float* __restrict__ rep, const float* __restrict__ winv,
    const float* __restrict__ binv, const float* __restrict__ x,
    const float* __restrict__ m, float* __restrict__ xp) {
  const int lane = threadIdx.x & 31, wave = threadIdx.x >> 5;
  const size_t r1 = (size_t)blockIdx.x * 8 + wave;
  float s = 0.0f;
#pragma unroll
  for (int j = 0; j < 8; ++j) {
    int e = lane + 32 * j;
    s += rep[r1 * 256 + e] * winv[e];
  }
  s = red_sum32(s);
  if (lane == 0) {
    int t = (int)(r1 >> 7);
    int bf = (int)(r1 & 127);
    int b = bf >> 4, f = bf & 15;
    size_t gi = ((size_t)b * 512 + t) * 16 + f;
    float xh = sigmoidf_(s + binv[0]);
    xp[gi] = m[gi] * x[gi] + (1.0f - m[gi]) * xh;
  }
}

// ---------------------------------------------------------------------------
// Stage 2 embedding: Xg = X'@wcomb^T + bcomb + PE ; tdf = exp(-relu(td@wtd^T+btd))
// row r2 = t*8 + b  (K = 16, VALU).
// ---------------------------------------------------------------------------
__global__ __launch_bounds__(256) void k_embed2(
    const float* __restrict__ xp, const float* __restrict__ wcomb,
    const float* __restrict__ bcomb, const float* __restrict__ td,
    const float* __restrict__ wtd, const float* __restrict__ btd,
    float* __restrict__ Xg, half_t* __restrict__ tdf) {
  size_t idx = (size_t)blockIdx.x * 256 + threadIdx.x;
  int e = (int)(idx & 255);
  size_t r = idx >> 8;
  int t = (int)(r >> 3), b = (int)(r & 7);
  size_t gi = ((size_t)b * 512 + t) * 16;
  float a1 = bcomb[e], a2 = btd[e];
#pragma unroll
  for (int f = 0; f < 16; ++f) {
    a1 += xp[gi + f] * wcomb[e * 16 + f];
    a2 += td[gi + f] * wtd[e * 16 + f];
  }
  int e2 = e & ~1;
  float div = __expf((float)e2 * (-9.210340371976184f / 256.0f));
  float ang = (float)t * div;
  a1 += (e & 1) ? __cosf(ang) : __sinf(ang);
  Xg[idx] = a1;
  tdf[idx] = (half_t)__expf(-fmaxf(a2, 0.0f));
}

// ---------------------------------------------------------------------------
// out[b,t,j] = sigmoid(rep2[t*8+b] . wout[j] + bout[j]). One wave per output.
// ---------------------------------------------------------------------------
__global__ __launch_bounds__(256) void k_out(
    const float* __restrict__ rep2, const float* __restrict__ wout,
    const float* __restrict__ bout, float* __restrict__ out) {
  const int lane = threadIdx.x & 31, wave = threadIdx.x >> 5;
  const size_t gid = (size_t)blockIdx.x * 8 + wave;  // flat (B,T,F) index
  int b = (int)(gid >> 13);
  int rem = (int)(gid & 8191);
  int t = rem >> 4, j = rem & 15;
  size_t row = (size_t)t * 8 + b;
  float s = 0.0f;
#pragma unroll
  for (int jj = 0; jj < 8; ++jj) {
    int e = lane + 32 * jj;
    s += rep2[row * 256 + e] * wout[(size_t)j * 256 + e];
  }
  s = red_sum32(s);
  if (lane == 0) out[gid] = sigmoidf_(s + bout[j]);
}

// ---------------------------------------------------------------------------
// Host orchestration
// ---------------------------------------------------------------------------
extern "C" void kernel_launch(void* const* d_in, const int* in_sizes, int n_in,
                              void* d_out, int out_size, void* d_ws, size_t ws_size,
                              hipStream_t stream) {
  (void)in_sizes; (void)n_in; (void)out_size; (void)ws_size;
  // d_in order: x, m, time_decay, then params in dict insertion order
  // (enc dicts expanded: wq,bq,wk,bk,wv,bv,wo,bo,w1,b1,w2,b2,g1,be1,g2,be2).
  const float* x    = (const float*)d_in[0];
  const float* m    = (const float*)d_in[1];
  const float* td   = (const float*)d_in[2];
  const float* w1x  = (const float*)d_in[3];
  const float* b1x  = (const float*)d_in[4];
  const float* w1m  = (const float*)d_in[5];
  const float* b1m  = (const float*)d_in[6];
  const float* winv = (const float*)d_in[7];
  const float* binv = (const float*)d_in[8];
  const float* wcomb = (const float*)d_in[25];
  const float* bcomb = (const float*)d_in[26];
  const float* wtd  = (const float*)d_in[43];
  const float* btd  = (const float*)d_in[44];
  const float* woutp = (const float*)d_in[45];
  const float* boutp = (const float*)d_in[46];
  float* outp = (float*)d_out;

  const int N1 = 65536;  // T * B*F = 512*128
  const int N2 = 4096;   // T * B   = 512*8
  char* ws = (char*)d_ws;
  // Large-phase buffers (with aliasing reuse; peak ~262 MB):
  float*  xe32   = (float*)(ws);                       // 67,108,864 B
  half_t* tde16  = (half_t*)(ws + 67108864);           // 33,554,432
  half_t* q16    = (half_t*)(ws + 100663296);          // 33,554,432
  half_t* k16    = (half_t*)(ws + 134217728);          // 33,554,432
  half_t* v16    = (half_t*)(ws + 167772160);          // 33,554,432
  half_t* ao16   = (half_t*)(ws + 201326592);          // 33,554,432
  float*  proj32 = (float*)(ws + 100663296);           // aliases q16+k16 (dead)
  float*  out132 = (float*)(ws + 167772160);           // aliases v16+ao16 (dead)
  float*  out232 = (float*)(ws);                       // aliases xe32 (dead)
  char* S = ws + 234881024;
  half_t* e1wq = (half_t*)(S);
  half_t* e1wk = (half_t*)(S + 131072);
  half_t* e1wv = (half_t*)(S + 262144);
  half_t* e1wo = (half_t*)(S + 393216);
  half_t* e1w1 = (half_t*)(S + 524288);
  half_t* e1w2 = (half_t*)(S + 1048576);
  half_t* e2wq = (half_t*)(S + 1572864);
  half_t* e2wk = (half_t*)(S + 1703936);
  half_t* e2wv = (half_t*)(S + 1835008);
  half_t* e2wo = (half_t*)(S + 1966080);
  half_t* e2w1 = (half_t*)(S + 2097152);
  half_t* e2w2 = (half_t*)(S + 2621440);
  char* S2 = S + 3145728;
  float*  xp    = (float*)(S2);
  float*  Xg32  = (float*)(S2 + 262144);
  half_t* tdf16 = (half_t*)(S2 + 4456448);
  half_t* q2    = (half_t*)(S2 + 6553600);
  half_t* k2    = (half_t*)(S2 + 8650752);
  half_t* v2    = (half_t*)(S2 + 10747904);
  half_t* ao2   = (half_t*)(S2 + 12845056);
  float*  proj2 = (float*)(S2 + 14942208);
  float*  o1_2  = (float*)(S2 + 19136512);
  float*  o2_2  = (float*)(S2 + 23330816);

  const int ATTN_LDS = (2 * 512 * 40 + 8 * 512) * 2;                 // 90,112 B
  const int FFN_LDS  = 32 * 258 * 2 + 32 * 1026 * 2 + 32 * 256 * 4;  // 114,944 B
  (void)hipFuncSetAttribute(reinterpret_cast<const void*>(k_attn),
                            hipFuncAttributeMaxDynamicSharedMemorySize, ATTN_LDS);
  (void)hipFuncSetAttribute(reinterpret_cast<const void*>(k_ffn),
                            hipFuncAttributeMaxDynamicSharedMemorySize, FFN_LDS);

  // Convert encoder weights to f16 once per launch (deterministic, ~3 MB).
  auto cvt = [&](int idx, half_t* dst, int n) {
    k_cvt<<<(n + 255) / 256, 256, 0, stream>>>((const float*)d_in[idx], dst, n);
  };
  cvt(9,  e1wq, 65536);  cvt(11, e1wk, 65536);
  cvt(13, e1wv, 65536);  cvt(15, e1wo, 65536);
  cvt(17, e1w1, 262144); cvt(19, e1w2, 262144);
  cvt(27, e2wq, 65536);  cvt(29, e2wk, 65536);
  cvt(31, e2wv, 65536);  cvt(33, e2wo, 65536);
  cvt(35, e2w1, 262144); cvt(37, e2w2, 262144);

  // ---- Stage 1: per-sensor embed + PE, time-decay embed ----
  k_embed1<<<65536, 256, 0, stream>>>(x, td, w1x, b1x, w1m, b1m, xe32, tde16);

  // ---- Encoder 1 (batch 128) ----
  dim3 g1(N1 / 64, 256 / 64);
  k_gemm<0><<<g1, 256, 0, stream>>>(xe32, e1wq, (const float*)d_in[10], nullptr, q16, 256, 256);
  k_gemm<0><<<g1, 256, 0, stream>>>(xe32, e1wk, (const float*)d_in[12], nullptr, k16, 256, 256);
  k_gemm<0><<<g1, 256, 0, stream>>>(xe32, e1wv, (const float*)d_in[14], nullptr, v16, 256, 256);
  k_attn<<<128 * 8, 256, ATTN_LDS, stream>>>(q16, k16, v16, ao16, 128);
  k_gemm<1><<<g1, 256, 0, stream>>>(ao16, e1wo, (const float*)d_in[16], proj32, nullptr, 256, 256);
  k_res_td_ln<<<N1 / 8, 256, 0, stream>>>(xe32, proj32, tde16,
      (const float*)d_in[21], (const float*)d_in[22], out132);
  k_ffn<<<N1 / 32, 256, FFN_LDS, stream>>>(out132, e1w1, (const float*)d_in[18],
      e1w2, (const float*)d_in[20], (const float*)d_in[23], (const float*)d_in[24], out232);

  // ---- Imputation + stage 2 embed ----
  k_xprime<<<N1 / 8, 256, 0, stream>>>(out232, winv, binv, x, m, xp);
  k_embed2<<<N2, 256, 0, stream>>>(xp, wcomb, bcomb, td, wtd, btd, Xg32, tdf16);

  // ---- Encoder 2 (batch 8) ----
  dim3 g2(N2 / 64, 256 / 64);
  k_gemm<0><<<g2, 256, 0, stream>>>(Xg32, e2wq, (const float*)d_in[28], nullptr, q2, 256, 256);
  k_gemm<0><<<g2, 256, 0, stream>>>(Xg32, e2wk, (const float*)d_in[30], nullptr, k2, 256, 256);
  k_gemm<0><<<g2, 256, 0, stream>>>(Xg32, e2wv, (const float*)d_in[32], nullptr, v2, 256, 256);
  k_attn<<<8 * 8, 256, ATTN_LDS, stream>>>(q2, k2, v2, ao2, 8);
  k_gemm<1><<<g2, 256, 0, stream>>>(ao2, e2wo, (const float*)d_in[34], proj2, nullptr, 256, 256);
  k_res_td_ln<<<N2 / 8, 256, 0, stream>>>(Xg32, proj2, tdf16,
      (const float*)d_in[39], (const float*)d_in[40], o1_2);
  k_ffn<<<N2 / 32, 256, FFN_LDS, stream>>>(o1_2, e2w1, (const float*)d_in[36],
      e2w2, (const float*)d_in[38], (const float*)d_in[41], (const float*)d_in[42], o2_2);

  // ---- Output projection ----
  k_out<<<(8 * 512 * 16) / 8, 256, 0, stream>>>(o2_2, woutp, boutp, outp);
}
